// KerasLSTMAttentionCell_52510270160994
// MI455X (gfx1250) — matrix operand order repaired and checked by reading, not compiled, not run on hardware
//
#include <hip/hip_runtime.h>
#include <hip/hip_bf16.h>

typedef __attribute__((ext_vector_type(16))) __bf16 v16bf;
typedef __attribute__((ext_vector_type(8)))  float  v8f;
typedef __attribute__((ext_vector_type(4)))  unsigned int u32x4;  // raw vec, no ctors
typedef unsigned short u16;
typedef unsigned int   u32;

// Problem constants
constexpr int LSTMH = 512, KMIX = 10, VOCABN = 80;
constexpr int BN = 64, TN = 512, UN = 64;

// Launch shape: 16 workgroups x 256 threads = 128 waves (wave32)
constexpr int NWG = 16, NTHR = 256;

// Padded K dims (multiples of 32 for WMMA k-tiles)
constexpr int K1PAD = 608,  K1T  = 19;  // [w(80)|x(3)|h1(512)] = 595 -> 608
constexpr int K23PAD = 1120, K23T = 35; // [x(3)|h(512)|w(80)|h(512)] = 1107 -> 1120

// ---- workspace layout (bytes) ----
constexpr size_t OFF_BAR   = 0;
constexpr size_t OFF_H1    = 256;
constexpr size_t SZ_STATE  = (size_t)BN * LSTMH * 4;
constexpr size_t OFF_C1    = OFF_H1 + SZ_STATE;
constexpr size_t OFF_H2    = OFF_C1 + SZ_STATE;
constexpr size_t OFF_C2    = OFF_H2 + SZ_STATE;
constexpr size_t OFF_H3    = OFF_C2 + SZ_STATE;
constexpr size_t OFF_C3    = OFF_H3 + SZ_STATE;
constexpr size_t OFF_KAPPA = OFF_C3 + SZ_STATE;
constexpr size_t OFF_WBUF  = OFF_KAPPA + (size_t)BN * KMIX * 4;
constexpr size_t OFF_SIN1  = OFF_WBUF + (size_t)BN * VOCABN * 4;   // bf16 [64][608]
constexpr size_t OFF_SIN2  = OFF_SIN1 + (size_t)BN * K1PAD * 2;    // bf16 [64][1120]
constexpr size_t OFF_SIN3  = OFF_SIN2 + (size_t)BN * K23PAD * 2;   // bf16 [64][1120]
constexpr size_t OFF_H1BF  = OFF_SIN3 + (size_t)BN * K23PAD * 2;   // bf16 [64][512], aligned h1n
constexpr size_t OFF_WAF   = OFF_H1BF + (size_t)BN * LSTMH * 2;    // packed wa frags
constexpr size_t SZ_WAF    = (size_t)K1T * 2 * 32 * 16 * 2;        // 38912 B
constexpr size_t OFF_W1F   = OFF_WAF + 77824;                       // rounded-up region
constexpr size_t SZ_W1F    = (size_t)K1T  * 128 * 32 * 16 * 2;
constexpr size_t OFF_W2F   = OFF_W1F + SZ_W1F;
constexpr size_t SZ_W23F   = (size_t)K23T * 128 * 32 * 16 * 2;
constexpr size_t OFF_W3F   = OFF_W2F + SZ_W23F;
constexpr size_t ZERO_U32  = OFF_W1F / 4;   // zero states + sins + h1bf (+waf, repacked anyway)

// ---- address-space(1) access helpers: force global_load/store, keep DScnt free ----
#define GAS __attribute__((address_space(1)))
__device__ __forceinline__ u32x4 ldg4(const u16* p) {
  return *reinterpret_cast<const GAS u32x4*>(reinterpret_cast<unsigned long long>(p));
}
__device__ __forceinline__ float ldg_f(const float* p) {
  return *reinterpret_cast<const GAS float*>(reinterpret_cast<unsigned long long>(p));
}
__device__ __forceinline__ void stg_f(float* p, float v) {
  *reinterpret_cast<GAS float*>(reinterpret_cast<unsigned long long>(p)) = v;
}
__device__ __forceinline__ void stg_h(u16* p, u16 v) {
  *reinterpret_cast<GAS u16*>(reinterpret_cast<unsigned long long>(p)) = v;
}

__device__ __forceinline__ u16 f2bf(float f) {  // RNE fp32 -> bf16
  u32 u = __float_as_uint(f);
  u32 r = u + 0x7fffu + ((u >> 16) & 1u);
  return (u16)(r >> 16);
}
__device__ __forceinline__ float sigm(float x) { return 1.0f / (1.0f + expf(-x)); }
__device__ __forceinline__ float softplusf(float x) {
  return (x > 20.0f) ? x : log1pf(expf(x));
}

union FragU {
  v16bf v;
  u32x4 q[2];
};

// Monotonic grid-wide barrier (16 co-resident workgroups)
__device__ __forceinline__ void grid_sync(u32* cnt, unsigned target) {
  __syncthreads();
  if (threadIdx.x == 0) {
    __threadfence();
    atomicAdd(cnt, 1u);
    while (__hip_atomic_load(cnt, __ATOMIC_RELAXED, __HIP_MEMORY_SCOPE_AGENT) < target)
      __builtin_amdgcn_s_sleep(1);
  }
  __syncthreads();
  __threadfence();
}

// Fused LSTM layer: z = sin @ Wcat (bf16 frags), 4 gates per wave, cell update.
// Wave owns (mtile 0..3, htile 0..31).
__device__ void lstm_layer(const u16* __restrict__ sa, int Kpad, int KT,
                           const u16* __restrict__ wf,
                           const float* __restrict__ bias,
                           float* __restrict__ hb, float* __restrict__ cb,
                           u16* __restrict__ nsin, int npitch,       // optional bf16 h out
                           u16* __restrict__ nsin2,                  // optional bf16 h out, pitch 512
                           float* __restrict__ outp, size_t ostride, // optional f32 h out
                           int mtile, int htile, int lane) {
  v8f acc0 = {}, acc1 = {}, acc2 = {}, acc3 = {};
  const int kb = (lane >> 4) * 8;
  const u16* arow = sa + (size_t)(mtile * 16 + (lane & 15)) * Kpad + kb;

  for (int kt = 0; kt < KT; ++kt) {
    FragU a;
    a.q[0] = ldg4(arow + kt * 32);
    a.q[1] = ldg4(arow + kt * 32 + 16);
    const u16* bbase = wf + (size_t)kt * (128 * 32 * 16) + (size_t)lane * 16;
    const u16* p0 = bbase + (size_t)(htile) * 512;
    const u16* p1 = bbase + (size_t)(32 + htile) * 512;
    const u16* p2 = bbase + (size_t)(64 + htile) * 512;
    const u16* p3 = bbase + (size_t)(96 + htile) * 512;
    FragU b0, b1, b2, b3;
    b0.q[0] = ldg4(p0); b0.q[1] = ldg4(p0 + 8);
    b1.q[0] = ldg4(p1); b1.q[1] = ldg4(p1 + 8);
    b2.q[0] = ldg4(p2); b2.q[1] = ldg4(p2 + 8);
    b3.q[0] = ldg4(p3); b3.q[1] = ldg4(p3 + 8);
    acc0 = __builtin_amdgcn_wmma_f32_16x16x32_bf16(false, a.v, false, b0.v, (short)0, acc0, false, false);
    acc1 = __builtin_amdgcn_wmma_f32_16x16x32_bf16(false, a.v, false, b1.v, (short)0, acc1, false, false);
    acc2 = __builtin_amdgcn_wmma_f32_16x16x32_bf16(false, a.v, false, b2.v, (short)0, acc2, false, false);
    acc3 = __builtin_amdgcn_wmma_f32_16x16x32_bf16(false, a.v, false, b3.v, (short)0, acc3, false, false);
  }

  const int hu = htile * 16 + (lane & 15);
  const float bi = ldg_f(bias + hu), bf_ = ldg_f(bias + LSTMH + hu);
  const float bg = ldg_f(bias + 2 * LSTMH + hu), bo = ldg_f(bias + 3 * LSTMH + hu);
  const int mbase = mtile * 16 + ((lane >> 4) << 3);
#pragma unroll
  for (int j = 0; j < 8; ++j) {
    const int m = mbase + j;
    const float zi = acc0[j] + bi;
    const float zf = acc1[j] + bf_;
    const float zg = acc2[j] + bg;
    const float zo = acc3[j] + bo;
    const float co = ldg_f(cb + m * LSTMH + hu);
    const float cn = sigm(zf) * co + sigm(zi) * tanhf(zg);
    const float hn = sigm(zo) * tanhf(cn);
    stg_f(cb + m * LSTMH + hu, cn);
    stg_f(hb + m * LSTMH + hu, hn);
    const u16 hv = f2bf(hn);
    if (nsin)  stg_h(nsin + (size_t)m * npitch + hu, hv);
    if (nsin2) stg_h(nsin2 + (size_t)m * LSTMH + hu, hv);
    if (outp)  stg_f(outp + (size_t)m * ostride + hu, hn);
  }
}

// Attention on wg0. Stage 1 is a WMMA GEMM: A = [w|x|pad|h1n] (sin1 cols 0..95 for
// kt<3 — cols 83..95 hold junk but multiply zero-padded wa rows — then h1bf), B = waf.
__device__ void attention_phase(int tid,
                                const float* __restrict__ av,
                                const float* __restrict__ ba,
                                const int* __restrict__ c_len,
                                const u16* __restrict__ sin1,
                                const u16* __restrict__ h1bf,
                                const u16* __restrict__ waf,
                                float* __restrict__ wbuf,
                                float* __restrict__ kappa_st,
                                u16* __restrict__ sin2, u16* __restrict__ sin3,
                                float* s_par, float* s_al, float* s_be, float* s_ka,
                                float* s_phi) {
  const int lane = tid & 31;
  // Stage 1: params = softplus([w|x|h1n] @ wa + ba) via 8 waves x 19 WMMAs
  {
    const int w8 = tid >> 5;   // 0..7
    const int amt = w8 & 3;    // mtile
    const int ant = w8 >> 2;   // 0..1 (N pad 30 -> 32)
    v8f pacc = {};
    const int kb = (lane >> 4) * 8;
    const u16* a1 = sin1 + (size_t)(amt * 16 + (lane & 15)) * K1PAD + kb;
    const u16* a2 = h1bf + (size_t)(amt * 16 + (lane & 15)) * LSTMH + kb;
    for (int kt = 0; kt < K1T; ++kt) {
      const u16* ap = (kt < 3) ? (a1 + kt * 32) : (a2 + (kt * 32 - 96));
      FragU a;
      a.q[0] = ldg4(ap);
      a.q[1] = ldg4(ap + 16);
      const u16* bp = waf + ((size_t)(kt * 2 + ant) * 32 + lane) * 16;
      FragU b;
      b.q[0] = ldg4(bp); b.q[1] = ldg4(bp + 8);
      pacc = __builtin_amdgcn_wmma_f32_16x16x32_bf16(false, a.v, false, b.v, (short)0, pacc, false, false);
    }
    const int c = ant * 16 + (lane & 15);
    const int mb = amt * 16 + ((lane >> 4) << 3);
    if (c < 30) {
      const float bav = ba[c];
#pragma unroll
      for (int j = 0; j < 8; ++j)
        s_par[(mb + j) * 30 + c] = softplusf(pacc[j] + bav);
    }
  }
  __syncthreads();
  // Stage 2: alpha/beta/kappa (64x10)
  for (int i = tid; i < BN * KMIX; i += NTHR) {
    const int b = i / KMIX, k = i - b * KMIX;
    const float alpha = s_par[b * 30 + k];
    const float beta  = fmaxf(s_par[b * 30 + 10 + k], 0.01f);
    float kap = kappa_st[i] + s_par[b * 30 + 20 + k] * 0.04f;
    kap = fminf(kap, (float)c_len[b] + 1.0f);
    kappa_st[i] = kap;
    s_al[i] = alpha; s_be[i] = beta; s_ka[i] = kap;
  }
  __syncthreads();
  // Stage 3: phi[b][u], masked by c_len
  for (int i = tid; i < BN * UN; i += NTHR) {
    const int b = i >> 6, u = i & 63;
    float phi = 0.f;
    if (u < c_len[b]) {
#pragma unroll
      for (int k = 0; k < KMIX; ++k) {
        const float d = s_ka[b * KMIX + k] - (float)u;
        phi += s_al[b * KMIX + k] * expf(-d * d / s_be[b * KMIX + k]);
      }
    }
    s_phi[i] = phi;
  }
  __syncthreads();
  // Stage 4: w[b][v] = sum_u phi * AV; publish f32 + bf16 into sin2/sin3
  for (int i = tid; i < BN * VOCABN; i += NTHR) {
    const int b = i / VOCABN, v = i - b * VOCABN;
    const float* avb = av + ((size_t)b * UN) * VOCABN + v;
    const float* ph  = s_phi + b * UN;
    float acc = 0.f;
    for (int u = 0; u < UN; ++u) acc = fmaf(ph[u], ldg_f(avb + (size_t)u * VOCABN), acc);
    stg_f(wbuf + b * VOCABN + v, acc);
    const u16 bv = f2bf(acc);
    stg_h(sin2 + b * K23PAD + 515 + v, bv);
    stg_h(sin3 + b * K23PAD + 515 + v, bv);
  }
}

// ---- persistent RNN kernel (4 grid barriers / timestep) ----
__global__ void __launch_bounds__(NTHR) rnn_persistent(
    const float* __restrict__ strokes, const float* __restrict__ av,
    const float* __restrict__ ba,
    const float* __restrict__ b1, const float* __restrict__ b2,
    const float* __restrict__ b3, const int* __restrict__ c_len,
    u32* bar,
    float* h1, float* c1, float* h2, float* c2, float* h3, float* c3,
    float* kappa_st, float* wbuf,
    u16* sin1, u16* sin2, u16* sin3, u16* h1bf,
    const u16* waf, const u16* w1f, const u16* w2f, const u16* w3f,
    float* out) {
  __shared__ float s_par[BN * 30];
  __shared__ float s_al[BN * KMIX], s_be[BN * KMIX], s_ka[BN * KMIX];
  __shared__ float s_phi[BN * UN];

  const int tid = threadIdx.x;
  const int wg = blockIdx.x;
  const int lane = tid & 31;
  const int gwave = wg * 8 + (tid >> 5);
  const int mtile = gwave & 3;
  const int htile = gwave >> 2;
  unsigned gen = 0;

  // Pre-loop: build sin1 for t=0 (w_prev = 0, h1 = 0 from zeroed ws)
  for (int i = wg * NTHR + tid; i < BN * 595; i += NWG * NTHR) {
    const int b = i / 595, k = i - b * 595;
    float v;
    if (k < 80)      v = ldg_f(wbuf + b * 80 + k);
    else if (k < 83) v = ldg_f(strokes + ((size_t)b * TN + 0) * 3 + (k - 80));
    else             v = ldg_f(h1 + b * LSTMH + (k - 83));
    stg_h(sin1 + b * K1PAD + k, f2bf(v));
  }
  gen += NWG; grid_sync(bar, gen);

  for (int t = 0; t < TN; ++t) {
    // Phase A: LSTM1 (emits h1n bf16 -> sin2 cols 3.. and h1bf),
    //          plus staging of sin2/sin3 x/h_prev regions (depend only on last step)
    lstm_layer(sin1, K1PAD, K1T, w1f, b1, h1, c1,
               sin2 + 3, K23PAD, h1bf, nullptr, 0, mtile, htile, lane);
    for (int i = wg * NTHR + tid; i < 2 * BN * 515; i += NWG * NTHR) {
      const int buf = i / (BN * 515);
      const int r = i - buf * (BN * 515);
      const int b = r / 515, k = r - b * 515;
      const float* hs = buf ? h3 : h2;
      u16* sp = buf ? sin3 : sin2;
      float v; int col;
      if (k < 3) { v = ldg_f(strokes + ((size_t)b * TN + t) * 3 + k); col = k; }
      else       { v = ldg_f(hs + b * LSTMH + (k - 3));               col = 595 + (k - 3); }
      stg_h(sp + b * K23PAD + col, f2bf(v));
    }
    gen += NWG; grid_sync(bar, gen);

    // Phase B: attention on wg0 (WMMA stage 1 + scalar window)
    if (wg == 0)
      attention_phase(tid, av, ba, c_len, sin1, h1bf, waf, wbuf, kappa_st,
                      sin2, sin3, s_par, s_al, s_be, s_ka, s_phi);
    gen += NWG; grid_sync(bar, gen);

    // Phase C: LSTM2 (emits h2n bf16 -> sin3 cols 3..)
    lstm_layer(sin2, K23PAD, K23T, w2f, b2, h2, c2,
               sin3 + 3, K23PAD, nullptr, nullptr, 0, mtile, htile, lane);
    gen += NWG; grid_sync(bar, gen);

    // Phase D: LSTM3 -> out[b][t][h], plus build sin1 for t+1
    lstm_layer(sin3, K23PAD, K23T, w3f, b3, h3, c3,
               nullptr, 0, nullptr, out + (size_t)t * LSTMH, (size_t)TN * LSTMH,
               mtile, htile, lane);
    if (t + 1 < TN) {
      for (int i = wg * NTHR + tid; i < BN * 595; i += NWG * NTHR) {
        const int b = i / 595, k = i - b * 595;
        float v;
        if (k < 80)      v = ldg_f(wbuf + b * 80 + k);
        else if (k < 83) v = ldg_f(strokes + ((size_t)b * TN + (t + 1)) * 3 + (k - 80));
        else             v = ldg_f(h1 + b * LSTMH + (k - 83));
        stg_h(sin1 + b * K1PAD + k, f2bf(v));
      }
    }
    gen += NWG; grid_sync(bar, gen);
  }
}

// ---- prep kernels ----
__global__ void zero_ws(u32* p, size_t n) {
  size_t i = (size_t)blockIdx.x * blockDim.x + threadIdx.x;
  const size_t stride = (size_t)gridDim.x * blockDim.x;
  for (; i < n; i += stride) p[i] = 0u;
}

// Pack concat(K[dK x 2048], R[512 x 2048], pad) into 32x16 bf16 B-fragments.
__global__ void pack_weights(const float* __restrict__ K, const float* __restrict__ R,
                             int dK, int KT, u16* __restrict__ frag) {
  const int total = KT * 128 * 32 * 16;
  for (int i = blockIdx.x * blockDim.x + threadIdx.x; i < total;
       i += gridDim.x * blockDim.x) {
    const int j = i & 15;
    const int lane = (i >> 4) & 31;
    const int nt = (i >> 9) & 127;
    const int kt = i >> 16;
    const int n = nt * 16 + (lane & 15);
    const int kg = kt * 32 + ((j >> 3) << 4) + ((lane >> 4) << 3) + (j & 7);
    float v = 0.f;
    if (kg < dK)            v = K[(size_t)kg * 2048 + n];
    else if (kg < dK + 512) v = R[(size_t)(kg - dK) * 2048 + n];
    frag[i] = f2bf(v);
  }
}

// Pack wa[595 x 30] into K-order [w(80)|x(3)|zero(13)|h(512)] x Npad(32) B-fragments.
__global__ void pack_wa(const float* __restrict__ wa, u16* __restrict__ frag) {
  const int total = K1T * 2 * 32 * 16;
  for (int i = blockIdx.x * blockDim.x + threadIdx.x; i < total;
       i += gridDim.x * blockDim.x) {
    const int j = i & 15;
    const int lane = (i >> 4) & 31;
    const int nt = (i >> 9) & 1;
    const int kt = i >> 10;
    const int n = nt * 16 + (lane & 15);
    const int kg = kt * 32 + ((j >> 3) << 4) + ((lane >> 4) << 3) + (j & 7);
    float v = 0.f;
    if (n < 30) {
      if (kg < 83)                    v = wa[(size_t)kg * 30 + n];
      else if (kg >= 96 && kg < 608)  v = wa[(size_t)(kg - 13) * 30 + n];
    }
    frag[i] = f2bf(v);
  }
}

extern "C" void kernel_launch(void* const* d_in, const int* in_sizes, int n_in,
                              void* d_out, int out_size, void* d_ws, size_t ws_size,
                              hipStream_t stream) {
  (void)in_sizes; (void)n_in; (void)out_size; (void)ws_size;
  const float* strokes = (const float*)d_in[0];
  const float* av      = (const float*)d_in[1];
  const float* k1      = (const float*)d_in[2];
  const float* r1      = (const float*)d_in[3];
  const float* b1      = (const float*)d_in[4];
  const float* wa      = (const float*)d_in[5];
  const float* ba      = (const float*)d_in[6];
  const float* k2      = (const float*)d_in[7];
  const float* r2      = (const float*)d_in[8];
  const float* b2      = (const float*)d_in[9];
  const float* k3      = (const float*)d_in[10];
  const float* r3      = (const float*)d_in[11];
  const float* b3      = (const float*)d_in[12];
  const int*   c_len   = (const int*)d_in[13];

  char* ws = (char*)d_ws;
  u32*   bar      = (u32*)(ws + OFF_BAR);
  float* h1       = (float*)(ws + OFF_H1);
  float* c1       = (float*)(ws + OFF_C1);
  float* h2       = (float*)(ws + OFF_H2);
  float* c2       = (float*)(ws + OFF_C2);
  float* h3       = (float*)(ws + OFF_H3);
  float* c3       = (float*)(ws + OFF_C3);
  float* kappa_st = (float*)(ws + OFF_KAPPA);
  float* wbuf     = (float*)(ws + OFF_WBUF);
  u16*   sin1     = (u16*)(ws + OFF_SIN1);
  u16*   sin2     = (u16*)(ws + OFF_SIN2);
  u16*   sin3     = (u16*)(ws + OFF_SIN3);
  u16*   h1bf     = (u16*)(ws + OFF_H1BF);
  u16*   waf      = (u16*)(ws + OFF_WAF);
  u16*   w1f      = (u16*)(ws + OFF_W1F);
  u16*   w2f      = (u16*)(ws + OFF_W2F);
  u16*   w3f      = (u16*)(ws + OFF_W3F);
  float* out      = (float*)d_out;

  zero_ws<<<256, NTHR, 0, stream>>>((u32*)d_ws, ZERO_U32);
  pack_wa<<<32, NTHR, 0, stream>>>(wa, waf);
  pack_weights<<<1024, NTHR, 0, stream>>>(k1, r1, 83,  K1T,  w1f);
  pack_weights<<<1024, NTHR, 0, stream>>>(k2, r2, 595, K23T, w2f);
  pack_weights<<<1024, NTHR, 0, stream>>>(k3, r3, 595, K23T, w3f);
  rnn_persistent<<<NWG, NTHR, 0, stream>>>(
      strokes, av, ba, b1, b2, b3, c_len, bar,
      h1, c1, h2, c2, h3, c3, kappa_st, wbuf,
      sin1, sin2, sin3, h1bf, waf, w1f, w2f, w3f, out);
}
